// QuestionModule_43198781063643
// MI455X (gfx1250) — compile-verified, matrix operand order checked by reading, not run on hardware
//
#include <hip/hip_runtime.h>
#include <hip/hip_bf16.h>

typedef __attribute__((ext_vector_type(16))) __bf16 v16bf;
typedef __attribute__((ext_vector_type(8)))  __bf16 v8bf;
typedef __attribute__((ext_vector_type(8)))  float  v8f;

// ---------------------------------------------------------------------------
// Fragment load matching CDNA5 16-bit A/B layout (CDNA5 ISA §7.12.2):
// lane L (row/col = L%16):
//   lanes 0-15 : elements 0..7 = K k0..k0+7,  8..15 = K k0+16..k0+23
//   lanes 16-31: elements 0..7 = K k0+8..k0+15, 8..15 = K k0+24..k0+31
// Caller passes k0 = kk + (lane>>4)*8, k1 = kk + 16 + (lane>>4)*8.
// Two contiguous 16B loads per lane -> global_load_b128-class traffic.
// ---------------------------------------------------------------------------
__device__ __forceinline__ v16bf load_frag(const __bf16* __restrict__ base,
                                           int k0, int k1) {
  v8bf lo = *(const v8bf*)(base + k0);
  v8bf hi = *(const v8bf*)(base + k1);
  v16bf r;
#pragma unroll
  for (int i = 0; i < 8; ++i) { r[i] = lo[i]; r[i + 8] = hi[i]; }
  return r;
}

// ---------------------------------------------------------------------------
// C[m][n] = sum_k A[m][k]*B[n][k] + bias[n]
// A: M x K row-major bf16, B: N x K row-major bf16 (weight matrix as stored),
// C: M x N row-major f32.  Block = 256 threads = 8 waves, each wave owns a
// 16(M) x 32(N) strip -> block covers 16 x 256.  K multiple of 32.
// ---------------------------------------------------------------------------
__global__ __launch_bounds__(256)
void wmma_gemm_bias(const __bf16* __restrict__ A,
                    const __bf16* __restrict__ Bm,
                    const float*  __restrict__ bias,
                    float*        __restrict__ C,
                    int M, int N, int K) {
  const int lane = threadIdx.x & 31;
  const int wave = threadIdx.x >> 5;
  const int half = lane >> 4;   // K-half selector per ISA layout
  const int l16  = lane & 15;

  const int mBase = blockIdx.x * 16;
  const int nBase = blockIdx.y * 256 + wave * 32;
  if (mBase >= M || nBase >= N) return;   // uniform per wave -> EXEC stays all-1s

  const __bf16* Arow  = A  + (size_t)(mBase + l16) * K;
  const __bf16* Brow0 = Bm + (size_t)(nBase + l16) * K;
  const __bf16* Brow1 = Bm + (size_t)(nBase + 16 + l16) * K;

  v8f acc0 = {};
  v8f acc1 = {};
  for (int kk = 0; kk < K; kk += 32) {
    const int k0 = kk + half * 8;
    const int k1 = kk + 16 + half * 8;
    v16bf a  = load_frag(Arow,  k0, k1);
    v16bf b0 = load_frag(Brow0, k0, k1);
    v16bf b1 = load_frag(Brow1, k0, k1);
    acc0 = __builtin_amdgcn_wmma_f32_16x16x32_bf16(false, a, false, b0,
                                                   (short)0, acc0, false, false);
    acc1 = __builtin_amdgcn_wmma_f32_16x16x32_bf16(false, a, false, b1,
                                                   (short)0, acc1, false, false);
  }

  const float bias0 = bias[nBase + l16];
  const float bias1 = bias[nBase + 16 + l16];
#pragma unroll
  for (int i = 0; i < 8; ++i) {
    const int m = mBase + half * 8 + i;      // C/D layout: VGPR i, half selects M 0-7/8-15
    C[(size_t)m * N + nBase + l16]       = acc0[i] + bias0;
    C[(size_t)m * N + nBase + 16 + l16]  = acc1[i] + bias1;
  }
}

// ---------------------------------------------------------------------------
// Embedding gather, fp32 -> bf16, reordered (B,S,E) -> (S,B,E) so the input
// projection is a plain row-major GEMM.
// ---------------------------------------------------------------------------
__global__ void gather_embed_bf16(const int* __restrict__ tokens,  // (B,S)
                                  const float* __restrict__ emb,   // (V,E)
                                  __bf16* __restrict__ xbf,        // (S,B,E)
                                  int B, int S, int E) {
  const int row = blockIdx.x;          // row = s*B + b
  const int s = row / B;
  const int b = row - s * B;
  const long tok = tokens[b * S + s];
  const float* __restrict__ src = emb + (size_t)tok * E;
  __bf16* __restrict__ dst = xbf + (size_t)row * E;
  for (int e = threadIdx.x; e < E; e += blockDim.x) dst[e] = (__bf16)src[e];
}

__global__ void f32_to_bf16_kernel(const float* __restrict__ src,
                                   __bf16* __restrict__ dst, int n) {
  const int i = blockIdx.x * blockDim.x + threadIdx.x;
  if (i < n) dst[i] = (__bf16)src[i];
}

__global__ void init_h_kernel(float* __restrict__ h32,
                              __bf16* __restrict__ hbf, int n) {
  const int i = blockIdx.x * blockDim.x + threadIdx.x;
  if (i < n) { h32[i] = 0.0f; hbf[i] = (__bf16)0.0f; }
}

// ---------------------------------------------------------------------------
// Fused GRU gates for one timestep:
//   r = sigmoid(xr+hr); z = sigmoid(xz+hz); n = tanh(xn + r*hn)
//   h = (1-z)*n + z*h_prev
// Writes fp32 h (state + optionally final output) and bf16 h for the next
// step's WMMA GEMM.
// ---------------------------------------------------------------------------
__global__ void gru_gates(const float* __restrict__ xp,    // (B,3H) slice at step s
                          const float* __restrict__ hp,    // (B,3H)
                          const float* __restrict__ hprev, // (B,H)
                          float*       __restrict__ hnew,  // (B,H)
                          __bf16*      __restrict__ hbf,   // (B,H)
                          float*       __restrict__ outp,  // d_out on last step
                          int Hd) {
  const int i = blockIdx.x * blockDim.x + threadIdx.x;
  const int b = i / Hd;
  const int j = i - b * Hd;
  const float* __restrict__ xrow = xp + (size_t)b * 3 * Hd;
  const float* __restrict__ hrow = hp + (size_t)b * 3 * Hd;

  const float r = 1.0f / (1.0f + __expf(-(xrow[j] + hrow[j])));
  const float z = 1.0f / (1.0f + __expf(-(xrow[j + Hd] + hrow[j + Hd])));
  const float n = tanhf(xrow[j + 2 * Hd] + r * hrow[j + 2 * Hd]);
  const float h = (1.0f - z) * n + z * hprev[i];

  hnew[i] = h;
  hbf[i]  = (__bf16)h;
  if (outp) outp[i] = h;
}

// ---------------------------------------------------------------------------
extern "C" void kernel_launch(void* const* d_in, const int* in_sizes, int n_in,
                              void* d_out, int out_size, void* d_ws, size_t ws_size,
                              hipStream_t stream) {
  (void)in_sizes; (void)n_in; (void)out_size; (void)ws_size;
  constexpr int B = 64, S = 128, H = 1024, E = 1024;
  constexpr int G = 3 * H;        // 3072
  constexpr int M1 = S * B;       // 8192 rows of the input projection

  const int*   tokens = (const int*)  d_in[0];
  const float* emb    = (const float*)d_in[1];
  const float* w_ih   = (const float*)d_in[2];
  const float* w_hh   = (const float*)d_in[3];
  const float* b_ih   = (const float*)d_in[4];
  const float* b_hh   = (const float*)d_in[5];

  // Workspace carve-up (256B aligned slices). Total ~131.5 MB.
  char* ws = (char*)d_ws;
  size_t off = 0;
  auto carve = [&](size_t bytes) -> void* {
    void* p = ws + off;
    off += (bytes + 255) & ~(size_t)255;
    return p;
  };
  __bf16* xbf   = (__bf16*)carve((size_t)M1 * E * 2);  // gathered embeddings, (S,B,E)
  __bf16* wihb  = (__bf16*)carve((size_t)G  * E * 2);  // w_ih bf16, (3H,E)
  __bf16* whhb  = (__bf16*)carve((size_t)G  * H * 2);  // w_hh bf16, (3H,H) — L2-resident
  float*  xproj = (float*) carve((size_t)M1 * G * 4);  // (S,B,3H)
  float*  hp    = (float*) carve((size_t)B  * G * 4);  // per-step h @ w_hh^T
  float*  hA    = (float*) carve((size_t)B  * H * 4);
  float*  hB    = (float*) carve((size_t)B  * H * 4);
  __bf16* hbf   = (__bf16*)carve((size_t)B  * H * 2);

  // 1) Weight conversion to bf16.
  f32_to_bf16_kernel<<<(G * E + 1023) / 1024, 1024, 0, stream>>>(w_ih, wihb, G * E);
  f32_to_bf16_kernel<<<(G * H + 1023) / 1024, 1024, 0, stream>>>(w_hh, whhb, G * H);

  // 2) Embedding gather + layout change (B,S,E) -> (S,B,E), bf16.
  gather_embed_bf16<<<M1, 256, 0, stream>>>(tokens, emb, xbf, B, S, E);

  // 3) Input projection: (8192x1024) x (1024x3072) + b_ih -> xproj.
  wmma_gemm_bias<<<dim3(M1 / 16, G / 256), 256, 0, stream>>>(
      xbf, wihb, b_ih, xproj, M1, G, E);

  // 4) h0 = 0.
  init_h_kernel<<<(B * H + 255) / 256, 256, 0, stream>>>(hA, hbf, B * H);

  // 5) Sequential recurrence: 128 dependent WMMA GEMMs + fused gates.
  float* hprev = hA;
  float* hnext = hB;
  for (int s = 0; s < S; ++s) {
    wmma_gemm_bias<<<dim3(B / 16, G / 256), 256, 0, stream>>>(
        hbf, whhb, b_hh, hp, B, G, H);
    float* outp = (s == S - 1) ? (float*)d_out : nullptr;
    gru_gates<<<(B * H + 255) / 256, 256, 0, stream>>>(
        xproj + (size_t)s * B * G, hp, hprev, hnext, hbf, outp, H);
    float* t = hprev; hprev = hnext; hnext = t;
  }
}
